// LinearSAE_73143293051550
// MI455X (gfx1250) — matrix operand order repaired and checked by reading, not compiled, not run on hardware
//
#include <hip/hip_runtime.h>
#include <stdint.h>

typedef __bf16 bf16;
typedef __attribute__((ext_vector_type(16))) __bf16 v16bf;
typedef __attribute__((ext_vector_type(8)))  __bf16 bf16x8;
typedef __attribute__((ext_vector_type(4)))  __bf16 bf16x4;
typedef __attribute__((ext_vector_type(8)))  float  v8f;
typedef __attribute__((ext_vector_type(4)))  float  f32x4;

#define D_MODEL   3072
#define D_SPARSE  24576
#define N_TOKENS  8192
#define K_SPARSE  128

#define BM  128
#define BN  128
#define BK  64
#define A_PITCH 72    // bf16 row pitch for A tiles (144B rows: 16B aligned, conflict-free)
#define B_PITCH 136   // bf16 row pitch for transposed B tiles (272B rows)

#define NTILES   (D_SPARSE / BN)   // 192
#define MTILES   (N_TOKENS / BM)   // 64
#define NCHUNK   48                // N-tiles per L2 chunk: W chunk 75MB + h 100MB < 192MB L2

// ---------------------------------------------------------------------------
// Kernel 0: bias2[s] = enc_bias[s] - dot(W_enc[s,:], pre_bias)
//   Folds (h - pre_bias)@W^T into h@W^T + bias2 so the GEMM inner loop is pure.
// ---------------------------------------------------------------------------
__global__ __launch_bounds__(256) void fold_bias_kernel(
    const float* __restrict__ W, const float* __restrict__ pre_bias,
    const float* __restrict__ enc_bias, float* __restrict__ bias2) {
  int s = blockIdx.x * 256 + threadIdx.x;
  const float* w = W + (size_t)s * D_MODEL;
  float acc = 0.0f;
  for (int d = 0; d < D_MODEL; d += 4) {
    f32x4 wv = *(const f32x4*)(w + d);
    f32x4 pv = *(const f32x4*)(pre_bias + d);
    acc += wv[0]*pv[0] + wv[1]*pv[1] + wv[2]*pv[2] + wv[3]*pv[3];
  }
  bias2[s] = enc_bias[s] - acc;
}

// ---------------------------------------------------------------------------
// Kernel 1: pre_acts = h @ W^T + bias2
//   bf16x3 split-precision WMMA (hi*hi + hi*lo + lo*hi ~ f32 accuracy).
//   Pipeline per K-stage:  async f32 tiles -> LDS  (overlaps compute)
//                          cooperative split to bf16 hi/lo (once per element)
//                          WMMA phase: b128 A fragment loads, per-element B loads.
// ---------------------------------------------------------------------------
__global__ __launch_bounds__(256, 1) void sae_gemm_kernel(
    const float* __restrict__ hp, const float* __restrict__ Wp,
    const float* __restrict__ bias2, float* __restrict__ out) {
  __shared__ float Af[BM * BK];          // raw f32 staging (async dest)  32KB
  __shared__ float Bf[BN * BK];          //                               32KB
  __shared__ bf16  Ahi[BM * A_PITCH];    // [m][k] bf16                   18KB
  __shared__ bf16  Alo[BM * A_PITCH];    //                               18KB
  __shared__ bf16  Bhi[BK * B_PITCH];    // [k][n] bf16 (transposed)      17KB
  __shared__ bf16  Blo[BK * B_PITCH];    //                               17KB

  const int tid  = threadIdx.x;
  const int lane = tid & 31;
  const int wave = tid >> 5;
  const int wm   = wave & 3;          // M offset = 32*wm
  const int wn   = wave >> 2;         // N offset = 64*wn
  const int l15  = lane & 15;
  const int lhi  = (lane >> 4) & 1;

  // L2-aware remap: 48 consecutive N-tiles x all 64 M-tiles per chunk
  // -> W_enc streamed from HBM exactly once (75MB W chunk + 100MB h < 192MB L2).
  const int blocksPerChunk = NCHUNK * MTILES;            // 3072
  const int bid   = blockIdx.x;
  const int chunk = bid / blocksPerChunk;
  const int rem   = bid % blocksPerChunk;
  const int nTile = chunk * NCHUNK + (rem % NCHUNK);
  const int mTile = rem / NCHUNK;

  const uint32_t mBase = (uint32_t)mTile * BM;
  const uint32_t nBase = (uint32_t)nTile * BN;

  const uint64_t hAddr = (uint64_t)hp;
  const uint64_t wAddr = (uint64_t)Wp;

  v8f acc[2][4];
#pragma unroll
  for (int t = 0; t < 2; ++t)
#pragma unroll
    for (int u = 0; u < 4; ++u)
#pragma unroll
      for (int r = 0; r < 8; ++r) acc[t][u][r] = 0.0f;

  // Async global->LDS staging of one 128x64 f32 tile of A and of B.
  // 2048 16B chunks per tile, 8 per thread per operand -> 16 async ops / wave.
  auto stage = [&](uint32_t kBase) {
    uint32_t aB = (uint32_t)(uintptr_t)&Af[0];
    uint32_t bB = (uint32_t)(uintptr_t)&Bf[0];
#pragma unroll
    for (int i = 0; i < 8; ++i) {
      uint32_t c  = (uint32_t)tid + (uint32_t)i * 256u;
      uint32_t r  = c >> 4;               // tile row
      uint32_t kc = (c & 15u) << 2;       // k offset in floats
      uint32_t goA = ((mBase + r) * (uint32_t)D_MODEL + kBase + kc) * 4u;
      uint32_t loA = aB + (r * BK + kc) * 4u;
      asm volatile("global_load_async_to_lds_b128 %0, %1, %2"
                   :: "v"(loA), "v"(goA), "s"(hAddr) : "memory");
      uint32_t goB = ((nBase + r) * (uint32_t)D_MODEL + kBase + kc) * 4u;
      uint32_t loB = bB + (r * BK + kc) * 4u;
      asm volatile("global_load_async_to_lds_b128 %0, %1, %2"
                   :: "v"(loB), "v"(goB), "s"(wAddr) : "memory");
    }
  };

  // Cooperative f32 -> (bf16 hi, bf16 lo) split; every element converted once.
  auto convertTiles = [&]() {
#pragma unroll
    for (int i = 0; i < 8; ++i) {           // A: keep [m][k], pack 4-wide stores
      int c = tid + i * 256;
      int m = c >> 4, kc = (c & 15) << 2;
      f32x4 x = *(const f32x4*)&Af[m * BK + kc];
      bf16x4 hv, lv;
#pragma unroll
      for (int j = 0; j < 4; ++j) {
        bf16 hb = (bf16)x[j];
        hv[j] = hb;
        lv[j] = (bf16)(x[j] - (float)hb);
      }
      *(bf16x4*)&Ahi[m * A_PITCH + kc] = hv;
      *(bf16x4*)&Alo[m * A_PITCH + kc] = lv;
    }
#pragma unroll
    for (int i = 0; i < 8; ++i) {           // B: transpose to [k][n]
      int c = tid + i * 256;
      int n = c >> 4, kc = (c & 15) << 2;
      f32x4 x = *(const f32x4*)&Bf[n * BK + kc];
#pragma unroll
      for (int j = 0; j < 4; ++j) {
        bf16 hb = (bf16)x[j];
        Bhi[(kc + j) * B_PITCH + n] = hb;
        Blo[(kc + j) * B_PITCH + n] = (bf16)(x[j] - (float)hb);
      }
    }
  };

  // Prologue: stage(0), convert it, start stage(1).
  stage(0);
  asm volatile("s_wait_asynccnt 0" ::: "memory");
  __syncthreads();
  convertTiles();
  __syncthreads();
  stage(BK);

  const int KSTAGES = D_MODEL / BK;   // 48
  for (int ks = 0; ks < KSTAGES; ++ks) {
    // ---- compute phase: bf16 buffers hold stage ks; async(ks+1) in flight ----
#pragma unroll
    for (int k0 = 0; k0 < BK; k0 += 32) {
      v16bf ahi[2], alo[2];
#pragma unroll
      for (int t = 0; t < 2; ++t) {
        // A 16x32 bf16 layout: lanes<16 hold K {0..7,16..23}, lanes>=16 {8..15,24..31}
        const bf16* ap = &Ahi[(wm * 32 + t * 16 + l15) * A_PITCH + k0 + lhi * 8];
        const bf16* al = &Alo[(wm * 32 + t * 16 + l15) * A_PITCH + k0 + lhi * 8];
        bf16x8 h0 = *(const bf16x8*)ap;
        bf16x8 h1 = *(const bf16x8*)(ap + 16);
        bf16x8 l0 = *(const bf16x8*)al;
        bf16x8 l1 = *(const bf16x8*)(al + 16);
        ahi[t] = __builtin_shufflevector(h0, h1, 0,1,2,3,4,5,6,7,8,9,10,11,12,13,14,15);
        alo[t] = __builtin_shufflevector(l0, l1, 0,1,2,3,4,5,6,7,8,9,10,11,12,13,14,15);
      }
#pragma unroll
      for (int u = 0; u < 4; ++u) {
        // B 32x16: lane = N column, vector elems = K {0..7,16..23} (lanes<16) / {8..15,24..31} (lanes>=16)
        const int ncol = wn * 64 + u * 16 + l15;
        v16bf bhi, blo;
#pragma unroll
        for (int s = 0; s < 16; ++s) {
          const int krow = k0 + lhi * 8 + (s < 8 ? s : 16 + (s - 8));
          bhi[s] = Bhi[krow * B_PITCH + ncol];
          blo[s] = Blo[krow * B_PITCH + ncol];
        }
#pragma unroll
        for (int t = 0; t < 2; ++t) {
          acc[t][u] = __builtin_amdgcn_wmma_f32_16x16x32_bf16(
              false, ahi[t], false, bhi, (short)0, acc[t][u], false, false);
          acc[t][u] = __builtin_amdgcn_wmma_f32_16x16x32_bf16(
              false, ahi[t], false, blo, (short)0, acc[t][u], false, false);
          acc[t][u] = __builtin_amdgcn_wmma_f32_16x16x32_bf16(
              false, alo[t], false, bhi, (short)0, acc[t][u], false, false);
        }
      }
    }
    // ---- pipeline tail: convert stage ks+1, then start async for ks+2 ----
    if (ks + 1 < KSTAGES) {
      asm volatile("s_wait_asynccnt 0" ::: "memory");  // my 16 ops for ks+1 done
      __syncthreads();                                 // everyone done: f32 tiles full, bf16 free
      convertTiles();
      __syncthreads();                                 // bf16 visible, f32 staging free
      if (ks + 2 < KSTAGES) stage((uint32_t)(ks + 2) * BK);
    }
  }

  // Epilogue: D layout -> lane N = lane&15, M = 8*(lane>>4) + r. Streaming NT stores
  // (805MB output exceeds L2; top-k pass re-reads it row-resident anyway).
#pragma unroll
  for (int u = 0; u < 4; ++u) {
    uint32_t col = nBase + wn * 64 + u * 16 + l15;
    float bb = bias2[col];
#pragma unroll
    for (int t = 0; t < 2; ++t) {
      uint32_t row0 = mBase + wm * 32 + t * 16 + lhi * 8;
#pragma unroll
      for (int r = 0; r < 8; ++r) {
        __builtin_nontemporal_store(acc[t][u][r] + bb,
                                    &out[(size_t)(row0 + r) * D_SPARSE + col]);
      }
    }
  }
}

// ---------------------------------------------------------------------------
// Kernel 2: per-row exact top-K threshold (radix rank-select in LDS) + ReLU mask
//   Whole 96KB row lives in the 320KB WGP LDS: 1 global read + 1 write per elem.
// ---------------------------------------------------------------------------
__global__ __launch_bounds__(256) void topk_mask_kernel(float* __restrict__ out) {
  extern __shared__ float srow[];                 // 24576 floats = 96 KB
  __shared__ unsigned int hist[256];
  __shared__ unsigned int sPrefix, sRem;
  const int tid = threadIdx.x;
  float* rowp = out + (size_t)blockIdx.x * D_SPARSE;

  for (int i = tid; i < D_SPARSE / 4; i += 256)
    ((f32x4*)srow)[i] = ((const f32x4*)rowp)[i];
  if (tid == 0) { sPrefix = 0u; sRem = K_SPARSE; }
  __syncthreads();

  for (int p = 0; p < 4; ++p) {
    hist[tid] = 0u;
    __syncthreads();
    const unsigned int prefix = sPrefix;
    const int shift = 24 - 8 * p;
    for (int i = tid; i < D_SPARSE; i += 256) {
      unsigned int u = __float_as_uint(srow[i]);
      u = (u & 0x80000000u) ? ~u : (u | 0x80000000u);   // order-preserving key
      bool ok;
      if (p == 0) ok = true;
      else        ok = ((u >> (shift + 8)) == prefix);
      if (ok) atomicAdd(&hist[(u >> shift) & 255u], 1u);
    }
    __syncthreads();
    if (tid == 0) {
      unsigned int rem = sRem;
      int d = 255;
      while (d > 0) {
        unsigned int c = hist[d];
        if (c >= rem) break;
        rem -= c;
        --d;
      }
      sRem = rem;
      sPrefix = (sPrefix << 8) | (unsigned int)d;
    }
    __syncthreads();
  }
  const unsigned int T = sPrefix;   // bit pattern of the K-th largest key
  for (int i = tid; i < D_SPARSE; i += 256) {
    float v = srow[i];
    unsigned int u = __float_as_uint(v);
    u = (u & 0x80000000u) ? ~u : (u | 0x80000000u);
    rowp[i] = (u >= T) ? fmaxf(v, 0.0f) : 0.0f;
  }
}

// ---------------------------------------------------------------------------
extern "C" void kernel_launch(void* const* d_in, const int* in_sizes, int n_in,
                              void* d_out, int out_size, void* d_ws, size_t ws_size,
                              hipStream_t stream) {
  (void)in_sizes; (void)n_in; (void)out_size; (void)ws_size;
  const float* h        = (const float*)d_in[0];
  const float* W        = (const float*)d_in[1];
  const float* pre_bias = (const float*)d_in[2];
  const float* enc_bias = (const float*)d_in[3];
  float* out   = (float*)d_out;
  float* bias2 = (float*)d_ws;   // 24576 floats = 96 KB scratch

  fold_bias_kernel<<<D_SPARSE / 256, 256, 0, stream>>>(W, pre_bias, enc_bias, bias2);

  sae_gemm_kernel<<<NTILES * MTILES, 256, 0, stream>>>(h, W, bias2, out);

  topk_mask_kernel<<<N_TOKENS, 256, (size_t)D_SPARSE * sizeof(float), stream>>>(out);
}
